// FeatureExtractor_32461362823481
// MI455X (gfx1250) — matrix operand order repaired and checked by reading, not compiled, not run on hardware
//
#include <hip/hip_runtime.h>
#include <hip/hip_bf16.h>
#include <math.h>

// ---------------------------------------------------------------------------
// MI455X (gfx1250) implementation. wave32; all matmuls via
// v_wmma_f32_16x16x32_f16 (f16 inputs, f32 accumulate).
// The GEMM double-buffers its B tile in LDS with global_load_async_to_lds_b128
// (ASYNCcnt / s_wait_asynccnt), prefetches the K-chunk after next with
// global_prefetch, loads A fragments as two b128 reads, and register-blocks
// 4 M-tiles per wave (4 back-to-back WMMAs per K-chunk).
// Conv biases are dropped: training-mode BN makes per-channel pre-BN bias
// a mathematical no-op.
// ---------------------------------------------------------------------------

typedef __attribute__((ext_vector_type(16))) _Float16 v16h;
typedef __attribute__((ext_vector_type(8)))  _Float16 v8h;
typedef __attribute__((ext_vector_type(8)))  float    v8f;
using f16 = _Float16;

#define N_NODES 2048
#define T_IN    512

// ---------------- fragment K maps (kmapA: ISA table order; kmapB: contiguous k = 16*hl + e, unused) --------
__device__ __forceinline__ int kmapA(int e, int hl) {           // 16-bit A 16x32
  return ((e < 8) ? e : (e + 8)) + 8 * hl;
}
__device__ __forceinline__ int kmapB(int e, int hl) {           // 16-bit B 32x16
  return e + 16 * hl;
}

// ---------------- CDNA5 async global->LDS helpers ---------------------------
__device__ __forceinline__ unsigned lds_off(const void* p) {
  // generic -> addrspace(3) -> 32-bit LDS byte offset
  return (unsigned)(unsigned long)(__attribute__((address_space(3))) const char*)p;
}
__device__ __forceinline__ void async_g2l_b128(const void* g, unsigned loff) {
  asm volatile("global_load_async_to_lds_b128 %0, %1, off" :: "v"(loff), "v"(g) : "memory");
}
__device__ __forceinline__ void wait_async2() {   // previous stage done
  asm volatile("s_wait_asynccnt 2" ::: "memory");
}
__device__ __forceinline__ void wait_async0() {   // all stages done
  asm volatile("s_wait_asynccnt 0" ::: "memory");
}

// ---------------------------------------------------------------------------
// LDS-staged, double-buffered, register-blocked WMMA GEMM:
//   C[M,N] = A[M,K] x B[K,N], row-major f16 in, f32 out.
// blockDim.x == 256 (8 waves): block computes a 64(M) x 128(N) tile; each
// wave owns a 16-col slice and 4 stacked 16x16 C tiles.
// grid = (N/128, M/64, Z); per-z strides (elements) batch LSTM directions.
// Requirements: K % 32 == 0, N % 128 == 0, M % 64 == 0, lda % 8 == 0,
// ldb even, 16B-aligned bases.
// ---------------------------------------------------------------------------
__global__ void __launch_bounds__(256)
gemm_lds(const f16* __restrict__ A, long lda, long zsa,
         const f16* __restrict__ B, long ldb, long zsb,
         float* __restrict__ C, long ldc, long zsc, int K) {
  __shared__ __align__(16) f16 tile[2][32 * 128];   // double-buffered B chunk
  int tid  = threadIdx.x;
  int lane = tid & 31;
  int wv   = tid >> 5;
  int hl   = lane >> 4;
  int l16  = lane & 15;
  int m0   = blockIdx.y * 64;
  long nblk = (long)blockIdx.x * 128;
  int z    = blockIdx.z;

  const f16* Az = A + (long)z * zsa;
  const f16* Bz = B + (long)z * zsb;
  float*     Cz = C + (long)z * zsc;

  // stage one 32x128 B chunk (k-rows k0..k0+31) into tile[buf]
  auto stage = [&](int k0, int buf) {
#pragma unroll
    for (int s = 0; s < 2; ++s) {
      int ch = tid + s * 256;          // 512 chunks of 8 halfs
      int kr = ch >> 4;                // 16 chunks per row
      int nc = (ch & 15) * 8;
      const f16* g = Bz + (long)(k0 + kr) * ldb + nblk + nc;
      async_g2l_b128(g, lds_off(&tile[buf][kr * 128 + nc]));
      if (k0 + 32 < K) __builtin_prefetch(g + 32 * ldb, 0, 1);
    }
  };

  int nch = K >> 5;
  stage(0, 0);

  v8f acc[4] = {};
  for (int c = 0; c < nch; ++c) {
    int cur = c & 1;
    if (c + 1 < nch) { stage((c + 1) << 5, cur ^ 1); wait_async2(); }
    else             { wait_async0(); }
    __syncthreads();

    int k0 = c << 5;

    // ---- B fragment from LDS (shared across the 4 M-tiles) ----
    v16h bv;
#pragma unroll
    for (int e = 0; e < 16; ++e)
      bv[e] = tile[cur][(((e < 8) ? e : (e + 8)) + 8 * hl) * 128 + wv * 16 + l16];

    // ---- 4 stacked A fragments, 4 WMMAs ----
#pragma unroll
    for (int mt = 0; mt < 4; ++mt) {
      const v8h* ap = (const v8h*)(Az + (long)(m0 + mt * 16 + l16) * lda + k0);
      v8h alo = ap[hl];        // K = k0 + 8*hl      .. +7
      v8h ahi = ap[2 + hl];    // K = k0 + 16 + 8*hl .. +7
      v16h av;
#pragma unroll
      for (int e = 0; e < 8; ++e) { av[e] = alo[e]; av[e + 8] = ahi[e]; }
      acc[mt] = __builtin_amdgcn_wmma_f32_16x16x32_f16(false, av, false, bv,
                                                       (short)0, acc[mt],
                                                       false, false);
    }
    __syncthreads();
  }

#pragma unroll
  for (int mt = 0; mt < 4; ++mt)
#pragma unroll
    for (int r = 0; r < 8; ++r)
      Cz[(long)(m0 + mt * 16 + r + 8 * hl) * ldc + nblk + wv * 16 + l16] = acc[mt][r];
}

// ---------------- small utility kernels ------------------------------------
__global__ void k_zero(float* p, long n) {
  long i = (long)blockIdx.x * 256 + threadIdx.x;
  if (i < n) p[i] = 0.0f;
}

__global__ void k_f32_to_f16(const float* in, f16* out, long n) {
  long i = (long)blockIdx.x * 256 + threadIdx.x;
  if (i < n) out[i] = (f16)in[i];
}

// conv weights [Cout][64][3] -> f16 [Cout][192], k = tap*64 + ic
__global__ void k_pack_convw(const float* w, f16* out, int Cout) {
  long i = (long)blockIdx.x * 256 + threadIdx.x;
  long total = (long)Cout * 192;
  if (i >= total) return;
  int c = (int)(i / 192);
  int k = (int)(i % 192);
  int tap = k / 64, ic = k % 64;
  out[i] = (f16)w[((long)c * 64 + ic) * 3 + tap];
}

// W [F][Kd] f32 -> W.T f16 [Kd][F]
__global__ void k_wT(const float* in, f16* out, int F, int Kd) {
  long i = (long)blockIdx.x * 256 + threadIdx.x;
  if (i >= (long)F * Kd) return;
  int f = (int)(i % F);
  int k = (int)(i / F);
  out[i] = (f16)in[(long)f * Kd + k];
}

// im2col -> B f16 [192, 2048*T].  layout==0: src = x [n][64][T]
//                                 layout==1: src = p1 [64][n][T]
__global__ void k_im2col(const float* src, f16* B, int T, int layout) {
  long i = (long)blockIdx.x * 256 + threadIdx.x;
  long NT = (long)N_NODES * T;
  long total = 192 * NT;
  if (i >= total) return;
  int  row = (int)(i / NT);
  long pos = i % NT;
  int tap = row / 64, ic = row % 64;
  int n = (int)(pos / T);
  int t = (int)(pos % T);
  int ts = t + tap - 1;
  float v = 0.0f;
  if (ts >= 0 && ts < T)
    v = (layout == 0) ? src[((long)n * 64 + ic) * T + ts]
                      : src[((long)ic * N_NODES + n) * T + ts];
  B[i] = (f16)v;
}

// per-channel batch stats over [n,t]:  y layout [C][2048][T], L = 2048*T
__global__ void k_bnstats(const float* y, float* mean, float* istd, long L) {
  __shared__ float s1[256], s2[256];
  long c = blockIdx.x;
  const float* row = y + c * L;
  float a = 0.0f, b = 0.0f;
  for (long i = threadIdx.x; i < L; i += 256) {
    float v = row[i]; a += v; b += v * v;
  }
  s1[threadIdx.x] = a; s2[threadIdx.x] = b;
  __syncthreads();
  for (int s = 128; s > 0; s >>= 1) {
    if (threadIdx.x < s) { s1[threadIdx.x] += s1[threadIdx.x + s]; s2[threadIdx.x] += s2[threadIdx.x + s]; }
    __syncthreads();
  }
  if (threadIdx.x == 0) {
    float m = s1[0] / (float)L;
    float v = s2[0] / (float)L - m * m;
    mean[c] = m;
    istd[c] = rsqrtf(v + 1e-5f);
  }
}

// BN -> ReLU -> maxpool2.  in y [C][2048][T] f32, out [C][2048][T/2]
template <typename TO>
__global__ void k_bnrp(const float* y, const float* mean, const float* istd,
                       const float* g, const float* bb, TO* out, int T, int C) {
  long i = (long)blockIdx.x * 256 + threadIdx.x;
  int T2 = T / 2;
  long total = (long)C * N_NODES * T2;
  if (i >= total) return;
  int t2 = (int)(i % T2);
  long r = i / T2;
  int n = (int)(r % N_NODES);
  int c = (int)(r / N_NODES);
  const float* src = y + ((long)c * N_NODES + n) * T + 2 * t2;
  float m = mean[c], is = istd[c], gg = g[c], bc = bb[c];
  float v0 = gg * (src[0] - m) * is + bc; v0 = v0 > 0.0f ? v0 : 0.0f;
  float v1 = gg * (src[1] - m) * is + bc; v1 = v1 > 0.0f ? v1 : 0.0f;
  out[i] = (TO)(v0 > v1 ? v0 : v1);
}

// p2 f16 [d][n][t] -> seqB f16 [d][t*2048+n]
__global__ void k_seqT(const f16* p2, f16* seqB) {
  long i = (long)blockIdx.x * 256 + threadIdx.x;
  if (i >= 33554432L) return;                  // 128 * 262144
  long d = i >> 18;
  long pos = i & 262143;
  int t = (int)(pos >> 11);
  int n = (int)(pos & 2047);
  seqB[i] = p2[(d << 18) + ((long)n << 7) + t];
}

// LSTM cell update (both directions), step t.
// pre_f/pre_r layout [g][t*2048+n] (both in forward time; dir 1 reads t'=127-t)
// gbuf [dir][g][n]; cst/hsum index i = dir*262144 + n*128 + hh;
// hstT written TRANSPOSED: [dir][hh][n] so the recurrent GEMM's B is
// row-major.
__global__ void k_cell(const float* pre_f, const float* pre_r, const float* gbuf,
                       const float* bf, const float* br,
                       float* cst, f16* hstT, float* hsum, int t) {
  int i = blockIdx.x * 256 + threadIdx.x;
  if (i >= 2 * N_NODES * 128) return;
  int hh  = i & 127;
  int n   = (i >> 7) & 2047;
  int dir = i >> 18;
  const float* pre  = dir ? pre_r : pre_f;
  const float* bias = dir ? br : bf;
  int teff = dir ? (127 - t) : t;
  long pbase = (long)teff * N_NODES + n;
  long gb    = (long)dir * 512 * N_NODES + n;
  float gi = pre[(long)(hh      ) * 262144 + pbase] + gbuf[gb + (long)(hh      ) * N_NODES] + bias[hh      ];
  float gf = pre[(long)(hh + 128) * 262144 + pbase] + gbuf[gb + (long)(hh + 128) * N_NODES] + bias[hh + 128];
  float gg = pre[(long)(hh + 256) * 262144 + pbase] + gbuf[gb + (long)(hh + 256) * N_NODES] + bias[hh + 256];
  float go = pre[(long)(hh + 384) * 262144 + pbase] + gbuf[gb + (long)(hh + 384) * N_NODES] + bias[hh + 384];
  float c  = cst[i];
  float si = 1.0f / (1.0f + expf(-gi));
  float sf = 1.0f / (1.0f + expf(-gf));
  float so = 1.0f / (1.0f + expf(-go));
  c = sf * c + si * tanhf(gg);
  float h = so * tanhf(c);
  cst[i] = c;
  hstT[(long)dir * 262144 + (long)hh * N_NODES + n] = (f16)h;
  hsum[i] += h;
}

// feats f16 [n][256]: first 128 = mean_t hf, last 128 = mean_t hr
__global__ void k_feats(const float* hsum, f16* feats) {
  int i = blockIdx.x * 256 + threadIdx.x;
  if (i >= N_NODES * 256) return;
  int k = i & 255;
  int n = i >> 8;
  float v = (k < 128) ? hsum[(long)n * 128 + k]
                      : hsum[262144 + (long)n * 128 + (k - 128)];
  feats[i] = (f16)(v * (1.0f / 128.0f));
}

// s1[i] = Wh[i,:] . a[0:128] ; s2[i] = Wh[i,:] . a[128:256]
__global__ void k_scores(const float* Wh, const float* a, float* s1, float* s2) {
  int n = blockIdx.x * 256 + threadIdx.x;
  if (n >= N_NODES) return;
  float x1 = 0.0f, x2 = 0.0f;
  for (int k = 0; k < 128; ++k) {
    float w = Wh[(long)n * 128 + k];
    x1 += w * a[k];
    x2 += w * a[128 + k];
  }
  s1[n] = x1;
  s2[n] = x2;
}

// per-row masked-softmax stats: rm = max, rl = sum exp(v - rm)
__global__ void k_smstats(const float* s1, const float* s2, const int* adj,
                          float* rm, float* rl) {
  __shared__ float sm[256];
  int i = blockIdx.x;
  int tid = threadIdx.x;
  float s1i = s1[i];
  float mx = -3.4e38f;
  for (int j = tid; j < N_NODES; j += 256) {
    float e = s1i + s2[j]; e = e > 0.0f ? e : 0.2f * e;
    float v = adj[(long)i * N_NODES + j] > 0 ? e : -9.0e15f;
    mx = fmaxf(mx, v);
  }
  sm[tid] = mx; __syncthreads();
  for (int s = 128; s > 0; s >>= 1) {
    if (tid < s) sm[tid] = fmaxf(sm[tid], sm[tid + s]);
    __syncthreads();
  }
  float m = sm[0];
  __syncthreads();
  float l = 0.0f;
  for (int j = tid; j < N_NODES; j += 256) {
    float e = s1i + s2[j]; e = e > 0.0f ? e : 0.2f * e;
    float v = adj[(long)i * N_NODES + j] > 0 ? e : -9.0e15f;
    l += expf(v - m);
  }
  sm[tid] = l; __syncthreads();
  for (int s = 128; s > 0; s >>= 1) {
    if (tid < s) sm[tid] += sm[tid + s];
    __syncthreads();
  }
  if (tid == 0) { rm[i] = m; rl[i] = sm[0]; }
}

// out[i, f] = sum_j softmax_row(i)[j] * Wh[j, f]; probabilities built per
// 16x32 fragment on the fly.  grid = (1, 2048/16), block 256.
__global__ void __launch_bounds__(256)
k_attgemm(const float* s1, const float* s2, const int* adj,
          const float* rm, const float* rl, const f16* Whh, float* out) {
  int lane = threadIdx.x & 31;
  int wv   = threadIdx.x >> 5;
  int hl   = lane >> 4;
  int l16  = lane & 15;
  int i0   = blockIdx.y * 16;
  int n0   = wv * 16;
  float s1r = s1[i0 + l16];
  float mr  = rm[i0 + l16];
  float inv_l = 1.0f / rl[i0 + l16];
  v8f acc = {};
  for (int kb = 0; kb < N_NODES; kb += 32) {
    v16h av, bv;
#pragma unroll
    for (int e = 0; e < 16; ++e) {
      int ka = kb + kmapA(e, hl);
      float ee = s1r + s2[ka]; ee = ee > 0.0f ? ee : 0.2f * ee;
      float v = adj[(long)(i0 + l16) * N_NODES + ka] > 0 ? ee : -9.0e15f;
      av[e] = (f16)(expf(v - mr) * inv_l);
      int kn = kb + kmapA(e, hl);
      bv[e] = Whh[(long)kn * 128 + n0 + l16];
    }
    acc = __builtin_amdgcn_wmma_f32_16x16x32_f16(false, av, false, bv,
                                                 (short)0, acc, false, false);
  }
#pragma unroll
  for (int r = 0; r < 8; ++r)
    out[(long)(i0 + r + 8 * hl) * 128 + n0 + l16] = acc[r];
}

__global__ void k_elu_f16(const float* in, f16* out, long n) {
  long i = (long)blockIdx.x * 256 + threadIdx.x;
  if (i >= n) return;
  float v = in[i];
  v = v > 0.0f ? v : (expf(v) - 1.0f);
  out[i] = (f16)v;
}

// ---------------------------------------------------------------------------
// workspace layout (bytes), all 16B-aligned
// ---------------------------------------------------------------------------
#define OFF_W1P    ((size_t)0)                 // f16 [64][192]
#define OFF_W2P    ((size_t)24576)             // f16 [128][192]
#define OFF_WIFH   ((size_t)73728)             // f16 [512][128]
#define OFF_WIRH   ((size_t)204800)            //   (adjacent: z-stride 65536 h)
#define OFF_WHFH   ((size_t)335872)            // f16 [512][128]
#define OFF_WHRH   ((size_t)466944)            //   (adjacent)
#define OFF_W1GT   ((size_t)598016)            // f16 [256][128]  (gat1_W^T)
#define OFF_W2GT   ((size_t)663552)            // f16 [128][128]  (gat2_W^T)
#define OFF_MEAN1  ((size_t)696320)
#define OFF_ISTD1  ((size_t)696576)
#define OFF_MEAN2  ((size_t)696832)
#define OFF_ISTD2  ((size_t)697344)
#define OFF_S1     ((size_t)697856)
#define OFF_S2     ((size_t)706048)
#define OFF_RM     ((size_t)714240)
#define OFF_RL     ((size_t)722432)
#define OFF_FEATS  ((size_t)730624)            // f16 [2048][256]
#define OFF_WH1    ((size_t)1779200)           // f32 [2048][128]
#define OFF_WH1H   ((size_t)2827776)           // f16 [2048][128]
#define OFF_ATT    ((size_t)3352064)           // f32 [2048][128]
#define OFF_H1F    ((size_t)4400640)           // f16 [2048][128]
#define OFF_WH2    ((size_t)4924928)           // f32 [2048][128]
#define OFF_WH2H   ((size_t)5973504)           // f16 [2048][128]
#define OFF_HST    ((size_t)6497792)           // f16 [2][128][2048]  (h^T)
#define OFF_CST    ((size_t)7546368)           // f32 [2][2048][128]
#define OFF_HSUM   ((size_t)9643520)           // f32 [2][2048][128]
#define OFF_GBUF   ((size_t)11740672)          // f32 [2][512][2048]
#define OFF_P2     ((size_t)20129280)          // f16 [128][2048][128]
#define OFF_BIG    ((size_t)87238144)          // transient region
// within BIG (phase 1: convs)
#define OFF_Y1     (OFF_BIG + 0)               // f32 [64][2048][512]
#define OFF_B1     (OFF_BIG + 268435456)       // f16 [192][1048576]
#define OFF_P1     (OFF_BIG + 268435456)       // f32 [64][2048][256] (B1 dead)
#define OFF_B2     (OFF_BIG + 402653184)       // f16 [192][524288]
#define OFF_Y2     (OFF_BIG + 603979776)       // f32 [128][2048][256]
// within BIG (phase 2: LSTM)
#define OFF_PREF   (OFF_BIG + 0)               // f32 [512][262144]
#define OFF_PRER   (OFF_BIG + 536870912)       // f32 [512][262144]
#define OFF_SEQB   (OFF_BIG + 1073741824)      // f16 [128][262144]

static inline dim3 ggrid(long M, long N, int Z) {
  return dim3((unsigned)(N / 128), (unsigned)(M / 64), (unsigned)Z);
}

extern "C" void kernel_launch(void* const* d_in, const int* in_sizes, int n_in,
                              void* d_out, int out_size, void* d_ws, size_t ws_size,
                              hipStream_t stream) {
  char* ws = (char*)d_ws;
  const float* x       = (const float*)d_in[0];
  const int*   adj     = (const int*)d_in[1];
  const float* conv1_w = (const float*)d_in[2];
  const float* bn1_g   = (const float*)d_in[4];
  const float* bn1_b   = (const float*)d_in[5];
  const float* conv2_w = (const float*)d_in[6];
  const float* bn2_g   = (const float*)d_in[8];
  const float* bn2_b   = (const float*)d_in[9];
  const float* wif     = (const float*)d_in[10];
  const float* whf     = (const float*)d_in[11];
  const float* bf      = (const float*)d_in[12];
  const float* wir     = (const float*)d_in[13];
  const float* whr     = (const float*)d_in[14];
  const float* br      = (const float*)d_in[15];
  const float* gat1_W  = (const float*)d_in[16];
  const float* gat1_a  = (const float*)d_in[17];
  const float* gat2_W  = (const float*)d_in[18];
  const float* gat2_a  = (const float*)d_in[19];
  float* out = (float*)d_out;

  f16*   w1p   = (f16*)(ws + OFF_W1P);
  f16*   w2p   = (f16*)(ws + OFF_W2P);
  f16*   wifh  = (f16*)(ws + OFF_WIFH);
  f16*   wirh  = (f16*)(ws + OFF_WIRH);
  f16*   whfh  = (f16*)(ws + OFF_WHFH);
  f16*   whrh  = (f16*)(ws + OFF_WHRH);
  f16*   w1gT  = (f16*)(ws + OFF_W1GT);
  f16*   w2gT  = (f16*)(ws + OFF_W2GT);
  float* mean1 = (float*)(ws + OFF_MEAN1);
  float* istd1 = (float*)(ws + OFF_ISTD1);
  float* mean2 = (float*)(ws + OFF_MEAN2);
  float* istd2 = (float*)(ws + OFF_ISTD2);
  float* s1    = (float*)(ws + OFF_S1);
  float* s2    = (float*)(ws + OFF_S2);
  float* rmv   = (float*)(ws + OFF_RM);
  float* rlv   = (float*)(ws + OFF_RL);
  f16*   feats = (f16*)(ws + OFF_FEATS);
  float* wh1   = (float*)(ws + OFF_WH1);
  f16*   wh1h  = (f16*)(ws + OFF_WH1H);
  float* att1  = (float*)(ws + OFF_ATT);
  f16*   h1f   = (f16*)(ws + OFF_H1F);
  float* wh2   = (float*)(ws + OFF_WH2);
  f16*   wh2h  = (f16*)(ws + OFF_WH2H);
  f16*   hstT  = (f16*)(ws + OFF_HST);
  float* cst   = (float*)(ws + OFF_CST);
  float* hsum  = (float*)(ws + OFF_HSUM);
  float* gbuf  = (float*)(ws + OFF_GBUF);
  f16*   p2    = (f16*)(ws + OFF_P2);
  float* y1    = (float*)(ws + OFF_Y1);
  f16*   b1    = (f16*)(ws + OFF_B1);
  float* p1    = (float*)(ws + OFF_P1);
  f16*   b2    = (f16*)(ws + OFF_B2);
  float* y2    = (float*)(ws + OFF_Y2);
  float* pre_f = (float*)(ws + OFF_PREF);
  float* pre_r = (float*)(ws + OFF_PRER);
  f16*   seqB  = (f16*)(ws + OFF_SEQB);

  // --- weight packing / f16 conversion ---
  k_pack_convw<<<(64 * 192 + 255) / 256, 256, 0, stream>>>(conv1_w, w1p, 64);
  k_pack_convw<<<(128 * 192 + 255) / 256, 256, 0, stream>>>(conv2_w, w2p, 128);
  k_f32_to_f16<<<(65536 + 255) / 256, 256, 0, stream>>>(wif, wifh, 65536);
  k_f32_to_f16<<<(65536 + 255) / 256, 256, 0, stream>>>(wir, wirh, 65536);
  k_f32_to_f16<<<(65536 + 255) / 256, 256, 0, stream>>>(whf, whfh, 65536);
  k_f32_to_f16<<<(65536 + 255) / 256, 256, 0, stream>>>(whr, whrh, 65536);
  k_wT<<<(32768 + 255) / 256, 256, 0, stream>>>(gat1_W, w1gT, 128, 256);
  k_wT<<<(16384 + 255) / 256, 256, 0, stream>>>(gat2_W, w2gT, 128, 128);

  // --- conv1: im2col + WMMA GEMM [64,192] x [192, 2048*512] ---
  {
    long N = (long)N_NODES * T_IN;
    k_im2col<<<(unsigned)((192 * N + 255) / 256), 256, 0, stream>>>(x, b1, T_IN, 0);
    gemm_lds<<<ggrid(64, N, 1), 256, 0, stream>>>(w1p, 192, 0, b1, N, 0, y1, N, 0, 192);
    k_bnstats<<<64, 256, 0, stream>>>(y1, mean1, istd1, N);
    long pt = 64L * N_NODES * 256;
    k_bnrp<float><<<(unsigned)((pt + 255) / 256), 256, 0, stream>>>(
        y1, mean1, istd1, bn1_g, bn1_b, p1, T_IN, 64);
  }

  // --- conv2: im2col + WMMA GEMM [128,192] x [192, 2048*256] ---
  {
    long N = (long)N_NODES * 256;
    k_im2col<<<(unsigned)((192 * N + 255) / 256), 256, 0, stream>>>(p1, b2, 256, 1);
    gemm_lds<<<ggrid(128, N, 1), 256, 0, stream>>>(w2p, 192, 0, b2, N, 0, y2, N, 0, 192);
    k_bnstats<<<128, 256, 0, stream>>>(y2, mean2, istd2, N);
    long pt = 128L * N_NODES * 128;
    k_bnrp<f16><<<(unsigned)((pt + 255) / 256), 256, 0, stream>>>(
        y2, mean2, istd2, bn2_g, bn2_b, p2, 256, 128);
  }

  // --- seq transpose + LSTM input projections (both dirs, one z=2 launch) ---
  k_seqT<<<131072, 256, 0, stream>>>(p2, seqB);
  gemm_lds<<<ggrid(512, 262144, 2), 256, 0, stream>>>(
      wifh, 128, 65536,          // A: z0 = W_if, z1 = W_ir (adjacent)
      seqB, 262144, 0,           // B shared
      pre_f, 262144, 134217728,  // C: z0 = pre_f, z1 = pre_r (both fwd time)
      128);

  // --- zero h^T, c, hsum (contiguous region) ---
  k_zero<<<(1310720 + 255) / 256, 256, 0, stream>>>((float*)(ws + OFF_HST), 1310720);

  // --- LSTM scan: 128 steps, both directions per step ---
  for (int t = 0; t < 128; ++t) {
    gemm_lds<<<ggrid(512, N_NODES, 2), 256, 0, stream>>>(
        whfh, 128, 65536,            // A: z0 = W_hf, z1 = W_hr
        hstT, N_NODES, 262144,       // B: h^T per dir, row-major [128][2048]
        gbuf, N_NODES, 1048576,      // C: [dir][512][2048]
        128);
    k_cell<<<(2 * N_NODES * 128 + 255) / 256, 256, 0, stream>>>(
        pre_f, pre_r, gbuf, bf, br, cst, hstT, hsum, t);
  }
  k_feats<<<(N_NODES * 256 + 255) / 256, 256, 0, stream>>>(hsum, feats);

  // --- GAT layer 1 ---
  gemm_lds<<<ggrid(N_NODES, 128, 1), 256, 0, stream>>>(
      feats, 256, 0, w1gT, 128, 0, wh1, 128, 0, 256);
  k_f32_to_f16<<<(262144 + 255) / 256, 256, 0, stream>>>(wh1, wh1h, 262144);
  k_scores<<<(N_NODES + 255) / 256, 256, 0, stream>>>(wh1, gat1_a, s1, s2);
  k_smstats<<<N_NODES, 256, 0, stream>>>(s1, s2, adj, rmv, rlv);
  k_attgemm<<<dim3(1, N_NODES / 16, 1), 256, 0, stream>>>(s1, s2, adj, rmv, rlv,
                                                          wh1h, att1);
  k_elu_f16<<<(262144 + 255) / 256, 256, 0, stream>>>(att1, h1f, 262144);

  // --- GAT layer 2 (writes d_out) ---
  gemm_lds<<<ggrid(N_NODES, 128, 1), 256, 0, stream>>>(
      h1f, 128, 0, w2gT, 128, 0, wh2, 128, 0, 128);
  k_f32_to_f16<<<(262144 + 255) / 256, 256, 0, stream>>>(wh2, wh2h, 262144);
  k_scores<<<(N_NODES + 255) / 256, 256, 0, stream>>>(wh2, gat2_a, s1, s2);
  k_smstats<<<N_NODES, 256, 0, stream>>>(s1, s2, adj, rmv, rlv);
  k_attgemm<<<dim3(1, N_NODES / 16, 1), 256, 0, stream>>>(s1, s2, adj, rmv, rlv,
                                                          wh2h, out);
}
